// Block_86028194939087
// MI455X (gfx1250) — matrix operand order repaired and checked by reading, not compiled, not run on hardware
//
#include <hip/hip_runtime.h>
#include <hip/hip_bf16.h>
#include <cstdint>

// ---------------------------------------------------------------------------
// Transformer block for MI455X (gfx1250, wave32, WMMA).
//   y = relu( (attn(x) @ Wo + bo) @ W1 + b1 ),  attn = causal MHA, H=16, Dh=64
// All GEMM-shaped math runs on v_wmma_f32_16x16x32_bf16 (fp32 accumulate).
// GEMM uses double-buffered LDS fed by async global->LDS copies (ASYNCcnt).
// ---------------------------------------------------------------------------

typedef __bf16 bf16;
typedef __attribute__((ext_vector_type(4)))  __bf16 v4bf;
typedef __attribute__((ext_vector_type(8)))  __bf16 v8bf;
typedef __attribute__((ext_vector_type(16))) __bf16 v16bf;
typedef __attribute__((ext_vector_type(8)))  float  v8f;

#define WMMA_BF16(a, b, c) \
  __builtin_amdgcn_wmma_f32_16x16x32_bf16(false, (a), false, (b), (short)0, (c), false, false)

static constexpr int Tt  = 1024;   // sequence length
static constexpr int Ee  = 1024;   // embed
static constexpr int Hh  = 16;     // heads
static constexpr int Dh  = 64;     // head size
static constexpr int Mm  = 8 * Tt; // B*T rows

#ifndef __has_builtin
#define __has_builtin(x) 0
#endif
#if __has_builtin(__builtin_amdgcn_global_load_async_to_lds_b128)
#define HAVE_ASYNC_LDS 1
#else
#define HAVE_ASYNC_LDS 0
#endif

// Builtin expects GCC-style int4 vector pointers in AS1 (global) / AS3 (LDS).
typedef int gvec4i __attribute__((vector_size(16)));
typedef __attribute__((address_space(1))) gvec4i as1_v4i;
typedef __attribute__((address_space(3))) gvec4i as3_v4i;

// 16-byte global -> LDS copy. Async (ASYNCcnt-tracked, no VGPR round trip)
// when the gfx1250 builtin is available; synchronous fallback otherwise.
// Flat->AS1: global flat address == AS1 address. Flat->AS3: LDS flat address
// keeps the byte offset in bits [31:0] (ISA 10.2 aperture rules).
static __device__ __forceinline__ void cp16(const bf16* g, bf16* s) {
#if HAVE_ASYNC_LDS
  __builtin_amdgcn_global_load_async_to_lds_b128(
      (as1_v4i*)(uintptr_t)g,
      (as3_v4i*)(unsigned)(uintptr_t)s, 0, 0);
#else
  *(uint4*)s = *(const uint4*)g;
#endif
}
static __device__ __forceinline__ void wait_async_le4() {
#if HAVE_ASYNC_LDS
#if __has_builtin(__builtin_amdgcn_s_wait_asynccnt)
  __builtin_amdgcn_s_wait_asynccnt(4);
#else
  asm volatile("s_wait_asynccnt 0x4" ::: "memory");
#endif
#endif
}
static __device__ __forceinline__ void wait_async_0() {
#if HAVE_ASYNC_LDS
#if __has_builtin(__builtin_amdgcn_s_wait_asynccnt)
  __builtin_amdgcn_s_wait_asynccnt(0);
#else
  asm volatile("s_wait_asynccnt 0x0" ::: "memory");
#endif
#endif
}

// Build a 16-element bf16 fragment from two 16-byte chunks (A/B VGPR layout:
// per lane, 8 bf16 in VGPRs 0-3 and 8 bf16 in VGPRs 4-7).
static __device__ __forceinline__ v16bf frag2(const bf16* lo, const bf16* hi) {
  v8bf a = *(const v8bf*)lo;
  v8bf b = *(const v8bf*)hi;
  return __builtin_shufflevector(a, b, 0,1,2,3,4,5,6,7,8,9,10,11,12,13,14,15);
}

// ---------------------------------------------------------------------------
// x (fp32) -> bf16, vectorized 4-wide
// ---------------------------------------------------------------------------
__global__ __launch_bounds__(256) void convert_x_kernel(const float* __restrict__ x,
                                                        bf16* __restrict__ xb) {
  int i = blockIdx.x * 256 + threadIdx.x;      // i indexes groups of 4
  float4 v = ((const float4*)x)[i];
  v4bf o;
  o[0] = (bf16)v.x; o[1] = (bf16)v.y; o[2] = (bf16)v.z; o[3] = (bf16)v.w;
  ((v4bf*)xb)[i] = o;
}

// ---------------------------------------------------------------------------
// W [k][n] fp32  ->  WT [n][k] bf16   (32x32 LDS tiles, 256 threads)
// ---------------------------------------------------------------------------
__global__ __launch_bounds__(256) void transpose_w_kernel(const float* __restrict__ W,
                                                          bf16* __restrict__ WT) {
  __shared__ float tile[32][33];
  int bx = blockIdx.x * 32;   // n base
  int by = blockIdx.y * 32;   // k base
  int tx = threadIdx.x & 31;
  int ty = threadIdx.x >> 5;  // 8 rows per pass
  #pragma unroll
  for (int r = ty; r < 32; r += 8)
    tile[r][tx] = W[(by + r) * Ee + bx + tx];          // W[k][n]
  __syncthreads();
  #pragma unroll
  for (int r = ty; r < 32; r += 8)
    WT[(bx + r) * Ee + by + tx] = (bf16)tile[tx][r];   // WT[n][k] = W[k][n]
}

// ---------------------------------------------------------------------------
// GEMM: C[8192 x 1024] = A[8192 x 1024] * BT[1024 x 1024]^T   (BT is [n][k])
// 128x128 block tile, 8 waves (4 in M x 2 in N), wave tile 32x64 = 8 WMMAs.
// Double-buffered LDS; async global->LDS prefetch one K-step ahead.
// Epilogue modes:
//   1: bf16 out scattered to [b][h][t][d]      (Q, K)
//   2: bf16 out scattered to [b][h][d][t]      (V transposed)
//   3: bf16 out row-major + bias               (out-proj -> FFN input)
//   4: f32  out row-major + bias + ReLU        (final output)
// ---------------------------------------------------------------------------
#define LSTR 40   // padded LDS row stride (bf16 elems) for 32-wide K tiles

__global__ __launch_bounds__(256) void gemm_bf16_kernel(
    const bf16* __restrict__ A, const bf16* __restrict__ BT,
    const float* __restrict__ bias,
    bf16* __restrict__ outB, float* __restrict__ outF, int mode) {
  __shared__ bf16 lA[2][128 * LSTR];
  __shared__ bf16 lB[2][128 * LSTR];

  int tid  = threadIdx.x;
  int lane = tid & 31;
  int wave = tid >> 5;
  int l15  = lane & 15;
  int half = lane >> 4;
  int bm = blockIdx.x * 128;
  int bn = blockIdx.y * 128;
  int wm = (wave & 3) * 32;   // wave row offset in tile
  int wn = (wave >> 2) * 64;  // wave col offset in tile

  // Straight-line staging: each thread owns 4 fixed 16B chunks per K-step
  // (rows r0 and r0+64 of the A and B tiles, at column c0).
  int r0 = tid >> 2;            // 0..63
  int c0 = (tid & 3) * 8;       // 0,8,16,24
  const bf16* gA0 = A  + (bm + r0) * Ee + c0;
  const bf16* gA1 = gA0 + 64 * Ee;
  const bf16* gB0 = BT + (bn + r0) * Ee + c0;
  const bf16* gB1 = gB0 + 64 * Ee;
  int sOff0 = r0 * LSTR + c0;
  int sOff1 = (r0 + 64) * LSTR + c0;

  // Prologue: stage K-step 0 into buffer 0.
  cp16(gA0, &lA[0][sOff0]);
  cp16(gA1, &lA[0][sOff1]);
  cp16(gB0, &lB[0][sOff0]);
  cp16(gB1, &lB[0][sOff1]);

  v8f acc[2][4] = {};
  const int NK = Ee / 32;
  for (int it = 0; it < NK; ++it) {
    int buf = it & 1;
    if (it + 1 < NK) {
      int kn = (it + 1) * 32;       // prefetch next K-step into other buffer
      cp16(gA0 + kn, &lA[buf ^ 1][sOff0]);
      cp16(gA1 + kn, &lA[buf ^ 1][sOff1]);
      cp16(gB0 + kn, &lB[buf ^ 1][sOff0]);
      cp16(gB1 + kn, &lB[buf ^ 1][sOff1]);
      wait_async_le4();             // older 4 copies (this buf) are done
    } else {
      wait_async_0();
    }
    __syncthreads();                // all waves' copies into buf visible

    v16bf af[2], bfv[4];
    #pragma unroll
    for (int i = 0; i < 2; ++i) {
      const bf16* p = &lA[buf][(wm + i * 16 + l15) * LSTR + half * 8];
      af[i] = frag2(p, p + 16);     // A: K = half*8 + {0..7}, +16
    }
    #pragma unroll
    for (int j = 0; j < 4; ++j) {
      const bf16* p = &lB[buf][(wn + j * 16 + l15) * LSTR + half * 8];
      bfv[j] = frag2(p, p + 16);    // B: K = half*8 + {0..7}, +16
    }
    #pragma unroll
    for (int i = 0; i < 2; ++i)
      #pragma unroll
      for (int j = 0; j < 4; ++j)
        acc[i][j] = WMMA_BF16(af[i], bfv[j], acc[i][j]);

    __syncthreads();                // everyone done reading buf before reuse
  }

  // Epilogue. C layout: VGPR g, lane l -> row = g + 8*(l>>4), col = l&15.
  #pragma unroll
  for (int i = 0; i < 2; ++i) {
    #pragma unroll
    for (int j = 0; j < 4; ++j) {
      #pragma unroll
      for (int g = 0; g < 8; ++g) {
        int row = bm + wm + i * 16 + g + half * 8;
        int col = bn + wn + j * 16 + l15;
        float v = acc[i][j][g];
        if (mode >= 3) v += bias[col];
        if (mode == 4) {
          outF[row * Ee + col] = v > 0.0f ? v : 0.0f;
        } else if (mode == 3) {
          outB[row * Ee + col] = (bf16)v;
        } else {
          int b = row >> 10, t = row & (Tt - 1);
          int h = col >> 6,  d = col & (Dh - 1);
          int idx = (mode == 1)
                      ? (((b * Hh + h) * Tt + t) * Dh + d)   // [b][h][t][d]
                      : (((b * Hh + h) * Dh + d) * Tt + t);  // [b][h][d][t]
          outB[idx] = (bf16)v;
        }
      }
    }
  }
}

// ---------------------------------------------------------------------------
// Flash-style causal attention. One wave per (b, h, 16-row q-tile).
// Keys processed 32 at a time: QK^T = 4 WMMAs, online softmax (shfl_xor
// reductions inside 16-lane halves), P re-striped via per-wave LDS into an
// A-fragment, PV = 4 WMMAs into 16x64 running output.
// ---------------------------------------------------------------------------
__global__ __launch_bounds__(256) void attn_kernel(
    const bf16* __restrict__ Qh,   // [b][h][t][d]
    const bf16* __restrict__ Kh,   // [b][h][t][d]
    const bf16* __restrict__ Vt,   // [b][h][d][t]
    bf16* __restrict__ Ob) {       // [b*T][E] bf16
  __shared__ bf16 P[8][16 * LSTR];  // per-wave P scratch (16 x 32, padded)

  int tid  = threadIdx.x;
  int lane = tid & 31;
  int wave = tid >> 5;
  int l15  = lane & 15;
  int half = lane >> 4;

  int task = blockIdx.x * 8 + wave;       // 8192 tasks total
  int b  = task >> 10;                    // / (H * 64)
  int rem = task & 1023;
  int h  = rem >> 6;
  int qt = rem & 63;
  int qb = qt * 16;

  const bf16* qbase = Qh + ((b * Hh + h) * Tt) * Dh;
  const bf16* kbase = Kh + ((b * Hh + h) * Tt) * Dh;
  const bf16* vbase = Vt + ((b * Hh + h) * Dh) * Tt;

  // Q fragments: A 16x64 split into two K=32 fragments.
  v16bf aq[2];
  {
    const bf16* qr = qbase + (qb + l15) * Dh;
    #pragma unroll
    for (int f = 0; f < 2; ++f) {
      int kk = f * 32 + half * 8;
      aq[f] = frag2(qr + kk, qr + kk + 16);
    }
  }

  v8f o[4] = {};
  float mrow[8], srow[8];
  #pragma unroll
  for (int g = 0; g < 8; ++g) { mrow[g] = -__builtin_inff(); srow[g] = 0.0f; }

  bf16* pl = &P[wave][0];
  const float scale = 0.125f;   // Dh^-0.5
  int kend = qb + 16;

  for (int kb = 0; kb < kend; kb += 32) {
    // ---- scores: S[16 x 32] = Q * K^T (two 16-col WMMA tiles, 2 K-steps) ----
    v8f s0 = {}, s1 = {};
    #pragma unroll
    for (int f = 0; f < 2; ++f) {
      const bf16* kr0 = kbase + (kb + l15) * Dh + f * 32 + half * 8;
      v16bf bk0 = frag2(kr0, kr0 + 16);
      const bf16* kr1 = kbase + (kb + 16 + l15) * Dh + f * 32 + half * 8;
      v16bf bk1 = frag2(kr1, kr1 + 16);
      s0 = WMMA_BF16(aq[f], bk0, s0);
      s1 = WMMA_BF16(aq[f], bk1, s1);
    }

    // ---- causal mask + online softmax ----
    float p0[8], p1[8];
    #pragma unroll
    for (int g = 0; g < 8; ++g) {
      int row = qb + g + half * 8;
      int c0  = kb + l15;
      float x0 = (c0      <= row) ? s0[g] * scale : -__builtin_inff();
      float x1 = (c0 + 16 <= row) ? s1[g] * scale : -__builtin_inff();
      float vmax = fmaxf(x0, x1);
      #pragma unroll
      for (int off = 1; off < 16; off <<= 1)
        vmax = fmaxf(vmax, __shfl_xor(vmax, off, 32));
      float mnew  = fmaxf(mrow[g], vmax);
      float e0    = __expf(x0 - mnew);
      float e1    = __expf(x1 - mnew);
      float alpha = __expf(mrow[g] - mnew);
      float rsum  = e0 + e1;
      #pragma unroll
      for (int off = 1; off < 16; off <<= 1)
        rsum += __shfl_xor(rsum, off, 32);
      srow[g] = srow[g] * alpha + rsum;
      mrow[g] = mnew;
      p0[g] = e0; p1[g] = e1;
      #pragma unroll
      for (int j = 0; j < 4; ++j) o[j][g] *= alpha;
    }

    // ---- re-stripe P (C layout) -> A fragment via per-wave LDS ----
    #pragma unroll
    for (int g = 0; g < 8; ++g) {
      int r = g + half * 8;
      pl[r * LSTR + l15]      = (bf16)p0[g];
      pl[r * LSTR + 16 + l15] = (bf16)p1[g];
    }
    asm volatile("s_wait_dscnt 0" ::: "memory");
    const bf16* pp = pl + l15 * LSTR + half * 8;
    v16bf ap = frag2(pp, pp + 16);

    // ---- O += P * V  (V^T is [d][t]: contiguous keys per d row) ----
    #pragma unroll
    for (int j = 0; j < 4; ++j) {
      const bf16* vr = vbase + (j * 16 + l15) * Tt + kb + half * 8;
      v16bf bv = frag2(vr, vr + 16);
      o[j] = WMMA_BF16(ap, bv, o[j]);
    }
  }

  // ---- normalize and store to [b*T][E] (col = h*64 + d) ----
  #pragma unroll
  for (int j = 0; j < 4; ++j) {
    #pragma unroll
    for (int g = 0; g < 8; ++g) {
      int row = qb + g + half * 8;
      int col = h * Dh + j * 16 + l15;
      Ob[(b * Tt + row) * Ee + col] = (bf16)(o[j][g] / srow[g]);
    }
  }
}

// ---------------------------------------------------------------------------
// Launcher. Workspace overlay (bytes):
//   xb   @ 0        16 MiB   bf16 x
//   W*T  @ 16 MiB   5 x 2 MiB bf16 transposed weights
//   Qh   @ 26 MiB   16 MiB
//   Kh   @ 42 MiB   16 MiB
//   Vt   @ 58 MiB   16 MiB
//   Ob   @ 74 MiB   16 MiB
//   Yb   @ 26 MiB   (reuses Qh after attention)       peak ~90 MiB
// ---------------------------------------------------------------------------
extern "C" void kernel_launch(void* const* d_in, const int* in_sizes, int n_in,
                              void* d_out, int out_size, void* d_ws, size_t ws_size,
                              hipStream_t stream) {
  (void)in_sizes; (void)n_in; (void)out_size; (void)ws_size;
  const float* x  = (const float*)d_in[0];
  const float* Wq = (const float*)d_in[1];
  const float* Wk = (const float*)d_in[2];
  const float* Wv = (const float*)d_in[3];
  const float* Wo = (const float*)d_in[4];
  const float* bo = (const float*)d_in[5];
  const float* W1 = (const float*)d_in[6];
  const float* b1 = (const float*)d_in[7];
  float* out = (float*)d_out;

  char* ws = (char*)d_ws;
  const size_t MB = 1024ull * 1024ull;
  bf16* xb  = (bf16*)(ws + 0);
  bf16* WqT = (bf16*)(ws + 16 * MB);
  bf16* WkT = (bf16*)(ws + 18 * MB);
  bf16* WvT = (bf16*)(ws + 20 * MB);
  bf16* WoT = (bf16*)(ws + 22 * MB);
  bf16* W1T = (bf16*)(ws + 24 * MB);
  bf16* Qh  = (bf16*)(ws + 26 * MB);
  bf16* Kh  = (bf16*)(ws + 42 * MB);
  bf16* Vt  = (bf16*)(ws + 58 * MB);
  bf16* Ob  = (bf16*)(ws + 74 * MB);
  bf16* Yb  = (bf16*)(ws + 26 * MB);   // reuse Qh region

  // 1) convert x to bf16
  convert_x_kernel<<<(Mm * Ee) / (4 * 256), 256, 0, stream>>>(x, xb);

  // 2) transpose weights to bf16 [n][k]
  dim3 tg(32, 32);
  transpose_w_kernel<<<tg, 256, 0, stream>>>(Wq, WqT);
  transpose_w_kernel<<<tg, 256, 0, stream>>>(Wk, WkT);
  transpose_w_kernel<<<tg, 256, 0, stream>>>(Wv, WvT);
  transpose_w_kernel<<<tg, 256, 0, stream>>>(Wo, WoT);
  transpose_w_kernel<<<tg, 256, 0, stream>>>(W1, W1T);

  // 3) projections (scatter into head layouts)
  dim3 gg(Mm / 128, Ee / 128);
  gemm_bf16_kernel<<<gg, 256, 0, stream>>>(xb, WqT, nullptr, Qh, nullptr, 1);
  gemm_bf16_kernel<<<gg, 256, 0, stream>>>(xb, WkT, nullptr, Kh, nullptr, 1);
  gemm_bf16_kernel<<<gg, 256, 0, stream>>>(xb, WvT, nullptr, Vt, nullptr, 2);

  // 4) causal attention
  attn_kernel<<<(8 * Hh * (Tt / 16)) / 8, 256, 0, stream>>>(Qh, Kh, Vt, Ob);

  // 5) output projection + bias -> bf16
  gemm_bf16_kernel<<<gg, 256, 0, stream>>>(Ob, WoT, bo, Yb, nullptr, 3);

  // 6) FFN: relu(Y @ W1 + b1) -> f32 out
  gemm_bf16_kernel<<<gg, 256, 0, stream>>>(Yb, W1T, b1, nullptr, out, 4);
}
